// MultiHeadedAttention_8366596293068
// MI455X (gfx1250) — matrix operand order repaired and checked by reading, not compiled, not run on hardware
//
#include <hip/hip_runtime.h>
#include <hip/hip_bf16.h>

typedef __attribute__((ext_vector_type(16))) _Float16 v16h_t;
typedef __attribute__((ext_vector_type(8)))  _Float16 v8h_t;
typedef __attribute__((ext_vector_type(8)))  float    v8f_t;

union Frag16 { v16h_t v; v8h_t h[2]; };

static __device__ __forceinline__ v8f_t wmma_f16(v16h_t a, v16h_t b, v8f_t c) {
  // (neg_a, A, neg_b, B, c_mod, C, reuse_a, reuse_b)
  return __builtin_amdgcn_wmma_f32_16x16x32_f16(false, a, false, b, (short)0, c, false, false);
}

constexpr int GM  = 4096;   // B*S rows
constexpr int GK  = 1024;   // inner dim
constexpr int GN  = 1024;   // model dim
constexpr int TM  = 128;    // block tile M
constexpr int TNB = 128;    // block tile N
constexpr int TK  = 32;     // k-step (WMMA K)
constexpr int LDK = 48;     // padded LDS row stride (halves): 96B keeps 16B align
constexpr int SEQ = 1024;
constexpr int NB  = 4;
constexpr int NH  = 16;
constexpr int DHD = 64;

enum { OUT_H16 = 0, OUT_H16_T = 1, OUT_F32 = 2 };

static __device__ __forceinline__ v8h_t cvt8(float4 a, float4 b) {
  v8h_t r;
  r[0] = (_Float16)a.x; r[1] = (_Float16)a.y; r[2] = (_Float16)a.z; r[3] = (_Float16)a.w;
  r[4] = (_Float16)b.x; r[5] = (_Float16)b.y; r[6] = (_Float16)b.z; r[7] = (_Float16)b.w;
  return r;
}

template <int MODE>
static __device__ __forceinline__ void store_elem(void* outp, int m, int n, float v) {
  if (MODE == OUT_F32) {
    ((float*)outp)[(size_t)m * GN + n] = v;
  } else if (MODE == OUT_H16) {
    ((_Float16*)outp)[(size_t)m * GN + n] = (_Float16)v;
  } else {  // transposed V: [(b*NH+h)*DHD + d][s]
    int bb = m >> 10, ss = m & 1023, hh = n >> 6, dd = n & 63;
    ((_Float16*)outp)[(size_t)((bb * NH + hh) * DHD + dd) * SEQ + ss] = (_Float16)v;
  }
}

// ---------------------------------------------------------------------------
// GEMM: out = (A[GMxGK] @ W[GNxGK]^T + bias) * scale
// 128x128 block tile, 8 waves in 2(M)x4(N), 64x32 wave tile (8 WMMA/k-step),
// register-staged software pipeline over the LDS tiles.
// ---------------------------------------------------------------------------
template <int MODE>
__global__ __launch_bounds__(256) void gemm_bias_kernel(
    const float* __restrict__ A, const float* __restrict__ W,
    const float* __restrict__ bias, void* __restrict__ outp, float scale) {
  __shared__ _Float16 lA[TM * LDK];
  __shared__ _Float16 lB[TNB * LDK];

  const int t    = threadIdx.x;
  const int lane = t & 31;
  const int w    = t >> 5;
  const int wm   = (w >> 2) * 64;   // wave M offset: 0 / 64
  const int wn   = (w & 3) * 32;    // wave N offset: 0/32/64/96
  const int lrow = lane & 15;
  const int hi   = (lane >> 4) & 1;
  const int kA   = hi * 16;         // A-fragment chunk base
  const int kB   = hi * 16;         // B-fragment chunk base

  const int mBlk = blockIdx.y * TM;
  const int nBlk = blockIdx.x * TNB;

  // cooperative fill: 128 rows x 32 cols per tile, 16 floats per thread
  const int fr = t >> 1;            // 0..127
  const int fc = (t & 1) * 16;      // 0 / 16

  const float* aSrc = A + (size_t)(mBlk + fr) * GK + fc;
  const float* wSrc = W + (size_t)(nBlk + fr) * GK + fc;

  float4 fa0, fa1, fa2, fa3, fw0, fw1, fw2, fw3;
  fa0 = *(const float4*)(aSrc + 0);
  fa1 = *(const float4*)(aSrc + 4);
  fa2 = *(const float4*)(aSrc + 8);
  fa3 = *(const float4*)(aSrc + 12);
  fw0 = *(const float4*)(wSrc + 0);
  fw1 = *(const float4*)(wSrc + 4);
  fw2 = *(const float4*)(wSrc + 8);
  fw3 = *(const float4*)(wSrc + 12);

  v8f_t acc[4][2];
#pragma unroll
  for (int g = 0; g < 4; ++g) { acc[g][0] = (v8f_t){}; acc[g][1] = (v8f_t){}; }

  for (int kb = 0; kb < GK; kb += TK) {
    __syncthreads();
    *(v8h_t*)&lA[fr * LDK + fc + 0] = cvt8(fa0, fa1);
    *(v8h_t*)&lA[fr * LDK + fc + 8] = cvt8(fa2, fa3);
    *(v8h_t*)&lB[fr * LDK + fc + 0] = cvt8(fw0, fw1);
    *(v8h_t*)&lB[fr * LDK + fc + 8] = cvt8(fw2, fw3);
    __syncthreads();

    if (kb + TK < GK) {  // stage next tile; overlaps the WMMAs below
      const float* an = aSrc + kb + TK;
      fa0 = *(const float4*)(an + 0);
      fa1 = *(const float4*)(an + 4);
      fa2 = *(const float4*)(an + 8);
      fa3 = *(const float4*)(an + 12);
      const float* wn_ = wSrc + kb + TK;
      fw0 = *(const float4*)(wn_ + 0);
      fw1 = *(const float4*)(wn_ + 4);
      fw2 = *(const float4*)(wn_ + 8);
      fw3 = *(const float4*)(wn_ + 12);
    }

    Frag16 b0, b1, af;
    b0.h[0] = *(const v8h_t*)&lB[(wn + lrow) * LDK + kB];
    b0.h[1] = *(const v8h_t*)&lB[(wn + lrow) * LDK + kB + 8];
    b1.h[0] = *(const v8h_t*)&lB[(wn + 16 + lrow) * LDK + kB];
    b1.h[1] = *(const v8h_t*)&lB[(wn + 16 + lrow) * LDK + kB + 8];
#pragma unroll
    for (int g = 0; g < 4; ++g) {
      af.h[0] = *(const v8h_t*)&lA[(wm + g * 16 + lrow) * LDK + kA];
      af.h[1] = *(const v8h_t*)&lA[(wm + g * 16 + lrow) * LDK + kA + 8];
      acc[g][0] = wmma_f16(af.v, b0.v, acc[g][0]);
      acc[g][1] = wmma_f16(af.v, b1.v, acc[g][1]);
    }
  }

  // epilogue: C layout row = base + r + hi*8, col = lane%16
  const int n0    = nBlk + wn + lrow;
  const int n1    = n0 + 16;
  const float bi0 = bias[n0];
  const float bi1 = bias[n1];
#pragma unroll
  for (int g = 0; g < 4; ++g) {
    const int mb = mBlk + wm + g * 16 + hi * 8;
#pragma unroll
    for (int r = 0; r < 8; ++r) {
      store_elem<MODE>(outp, mb + r, n0, (acc[g][0][r] + bi0) * scale);
      store_elem<MODE>(outp, mb + r, n1, (acc[g][1][r] + bi1) * scale);
    }
  }
}

// ---------------------------------------------------------------------------
// Flash attention: per wave 16 queries x DH=64, keys streamed in chunks of 32.
// Row sums of P computed by WMMA against an all-ones B fragment (5th
// accumulator) instead of lane shuffles; only the row-max uses a butterfly.
// Qh/Kh: f16 [B*S][D] (Q pre-scaled by 1/8). Vt: f16 [(b*NH+h)*DHD + d][S].
// mask: bool bytes [B][S][S], true = masked. O: f32 [B*S][D].
// ---------------------------------------------------------------------------
__global__ __launch_bounds__(256) void attn_kernel(
    const _Float16* __restrict__ Qh, const _Float16* __restrict__ Kh,
    const _Float16* __restrict__ Vt, const unsigned char* __restrict__ mask,
    float* __restrict__ O) {
  __shared__ _Float16 lP[8][16 * LDK];  // per-wave P tile (16 q x 32 keys)

  const int t    = threadIdx.x;
  const int lane = t & 31;
  const int w    = t >> 5;
  const int lrow = lane & 15;
  const int hi   = (lane >> 4) & 1;
  const int kA   = hi * 16;
  const int kB   = hi * 16;

  const int b  = blockIdx.z;
  const int h  = blockIdx.y;
  const int qb = blockIdx.x * 128 + w * 16;
  const size_t rowBase = (size_t)b * SEQ;

  // Q A-fragments (d-chunks 0..31 and 32..63)
  Frag16 q0, q1;
  {
    const _Float16* qp = Qh + (rowBase + qb + lrow) * (size_t)GN + h * DHD;
    q0.h[0] = *(const v8h_t*)(qp + kA);
    q0.h[1] = *(const v8h_t*)(qp + kA + 8);
    q1.h[0] = *(const v8h_t*)(qp + 32 + kA);
    q1.h[1] = *(const v8h_t*)(qp + 32 + kA + 8);
  }

  Frag16 onef;  // all-ones B fragment for row sums
#pragma unroll
  for (int i = 0; i < 8; ++i) {
    onef.h[0][i] = (_Float16)1.0f;
    onef.h[1][i] = (_Float16)1.0f;
  }

  float mrow[8];
#pragma unroll
  for (int r = 0; r < 8; ++r) mrow[r] = -3.0e38f;
  v8f_t o0 = {}, o1 = {}, o2 = {}, o3 = {}, o4 = {};  // o4 = row sums

  _Float16* Pw = &lP[w][0];

  for (int kb = 0; kb < SEQ; kb += 32) {
    // ---- scores: two 16-key tiles, K B-fragments straight from global
    v8f_t s0 = {}, s1 = {};
    {
      const _Float16* kp = Kh + (rowBase + kb + lrow) * (size_t)GN + h * DHD;
      Frag16 kf0, kf1;
      kf0.h[0] = *(const v8h_t*)(kp + kB);
      kf0.h[1] = *(const v8h_t*)(kp + kB + 8);
      kf1.h[0] = *(const v8h_t*)(kp + 32 + kB);
      kf1.h[1] = *(const v8h_t*)(kp + 32 + kB + 8);
      s0 = wmma_f16(q0.v, kf0.v, s0);
      s0 = wmma_f16(q1.v, kf1.v, s0);
      const _Float16* kp2 = kp + (size_t)16 * GN;  // keys +16
      kf0.h[0] = *(const v8h_t*)(kp2 + kB);
      kf0.h[1] = *(const v8h_t*)(kp2 + kB + 8);
      kf1.h[0] = *(const v8h_t*)(kp2 + 32 + kB);
      kf1.h[1] = *(const v8h_t*)(kp2 + 32 + kB + 8);
      s1 = wmma_f16(q0.v, kf0.v, s1);
      s1 = wmma_f16(q1.v, kf1.v, s1);
    }

    // ---- mask (true => -1e9)
    {
      const unsigned char* mp0 =
          mask + (rowBase + qb + hi * 8) * (size_t)SEQ + kb + lrow;
#pragma unroll
      for (int r = 0; r < 8; ++r) {
        const unsigned char* mp = mp0 + (size_t)r * SEQ;
        if (mp[0])  s0[r] = -1.0e9f;
        if (mp[16]) s1[r] = -1.0e9f;
      }
    }

    // ---- online softmax: row-max butterfly across the 16 lanes per half
#pragma unroll
    for (int r = 0; r < 8; ++r) {
      float mx = fmaxf(s0[r], s1[r]);
#pragma unroll
      for (int off = 1; off < 16; off <<= 1)
        mx = fmaxf(mx, __shfl_xor(mx, off, 32));
      float mnew  = fmaxf(mrow[r], mx);
      float alpha = __expf(mrow[r] - mnew);
      mrow[r]     = mnew;
      float p0    = __expf(s0[r] - mnew);
      float p1    = __expf(s1[r] - mnew);
      int row     = r + hi * 8;
      Pw[row * LDK + lrow]      = (_Float16)p0;
      Pw[row * LDK + 16 + lrow] = (_Float16)p1;
      o0[r] *= alpha; o1[r] *= alpha; o2[r] *= alpha; o3[r] *= alpha;
      o4[r] *= alpha;
    }

    // ---- P (A-frag, 16x32) from per-wave LDS; compiler handles DScnt waits
    Frag16 pf;
    pf.h[0] = *(const v8h_t*)&Pw[lrow * LDK + kA];
    pf.h[1] = *(const v8h_t*)&Pw[lrow * LDK + kA + 8];

    // row sums: P x ones, accumulated in the matrix pipe
    o4 = wmma_f16(pf.v, onef.v, o4);

    // ---- P @ V : V^T B-fragments, contiguous along keys
    const _Float16* vp = Vt + (size_t)((b * NH + h) * DHD) * SEQ + kb + kB;
    {
      Frag16 vf;
      const _Float16* v0p = vp + (size_t)(0 * 16 + lrow) * SEQ;
      vf.h[0] = *(const v8h_t*)(v0p); vf.h[1] = *(const v8h_t*)(v0p + 8);
      o0 = wmma_f16(pf.v, vf.v, o0);
      const _Float16* v1p = vp + (size_t)(1 * 16 + lrow) * SEQ;
      vf.h[0] = *(const v8h_t*)(v1p); vf.h[1] = *(const v8h_t*)(v1p + 8);
      o1 = wmma_f16(pf.v, vf.v, o1);
      const _Float16* v2p = vp + (size_t)(2 * 16 + lrow) * SEQ;
      vf.h[0] = *(const v8h_t*)(v2p); vf.h[1] = *(const v8h_t*)(v2p + 8);
      o2 = wmma_f16(pf.v, vf.v, o2);
      const _Float16* v3p = vp + (size_t)(3 * 16 + lrow) * SEQ;
      vf.h[0] = *(const v8h_t*)(v3p); vf.h[1] = *(const v8h_t*)(v3p + 8);
      o3 = wmma_f16(pf.v, vf.v, o3);
    }
  }

  // ---- normalize + store O (f32, [b*S+q][h*64+d]); o4[r] = row sum
#pragma unroll
  for (int r = 0; r < 8; ++r) {
    float inv   = 1.0f / o4[r];
    size_t base = (rowBase + qb + hi * 8 + r) * (size_t)GN + h * DHD + lrow;
    O[base + 0]  = o0[r] * inv;
    O[base + 16] = o1[r] * inv;
    O[base + 32] = o2[r] * inv;
    O[base + 48] = o3[r] * inv;
  }
}

// ---------------------------------------------------------------------------
extern "C" void kernel_launch(void* const* d_in, const int* in_sizes, int n_in,
                              void* d_out, int out_size, void* d_ws,
                              size_t ws_size, hipStream_t stream) {
  const float* query = (const float*)d_in[0];
  const float* key_  = (const float*)d_in[1];
  const float* value = (const float*)d_in[2];
  const unsigned char* mask = (const unsigned char*)d_in[3];  // bool bytes
  const float* Wq = (const float*)d_in[4];
  const float* bq = (const float*)d_in[5];
  const float* Wk = (const float*)d_in[6];
  const float* bk = (const float*)d_in[7];
  const float* Wv = (const float*)d_in[8];
  const float* bv = (const float*)d_in[9];
  const float* Wo = (const float*)d_in[10];
  const float* bo = (const float*)d_in[11];

  _Float16* Qh = (_Float16*)d_ws;                 // 8 MB
  _Float16* Kh = Qh + (size_t)GM * GN;            // 8 MB
  _Float16* Vt = Kh + (size_t)GM * GN;            // 8 MB (transposed V)
  float*    Oo = (float*)(Vt + (size_t)GM * GN);  // 16 MB

  dim3 ggrid(GN / TNB, GM / TM);  // (8, 32)
  gemm_bias_kernel<OUT_H16><<<ggrid, 256, 0, stream>>>(query, Wq, bq, Qh, 0.125f);
  gemm_bias_kernel<OUT_H16><<<ggrid, 256, 0, stream>>>(key_,  Wk, bk, Kh, 1.0f);
  gemm_bias_kernel<OUT_H16_T><<<ggrid, 256, 0, stream>>>(value, Wv, bv, Vt, 1.0f);

  attn_kernel<<<dim3(SEQ / 128, NH, NB), 256, 0, stream>>>(Qh, Kh, Vt, mask, Oo);

  gemm_bias_kernel<OUT_F32><<<ggrid, 256, 0, stream>>>(Oo, Wo, bo, d_out, 1.0f);
}
